// SelfAttention_85607288144082
// MI455X (gfx1250) — matrix operand order repaired and checked by reading, not compiled, not run on hardware
//
#include <hip/hip_runtime.h>
#include <hip/hip_bf16.h>
#include <cstdint>

// ---------------------------------------------------------------------------
// PVT-style spatial-reduction attention on MI455X (gfx1250, wave32, WMMA).
// All matmuls use v_wmma_f32_16x16x32_f16 (f16 operands, f32 accum).
// A-operands are staged in LDS with CDNA5 async-to-LDS DMA (ASYNCcnt) so the
// 8 waves of a block share one copy instead of 8x redundant global traffic.
// Attention is flash-style (streaming softmax); P is transposed C/D-layout ->
// A-layout through a per-wave LDS tile.
// ---------------------------------------------------------------------------

typedef __attribute__((ext_vector_type(16))) _Float16 v16h;
typedef __attribute__((ext_vector_type(8)))  _Float16 v8h;
typedef __attribute__((ext_vector_type(8)))  float    v8f;

#define WMMA_F16(a, b, c)                                                     \
  __builtin_amdgcn_wmma_f32_16x16x32_f16(false, (a), false, (b), (short)0,    \
                                         (c), false, false)

static constexpr int kB  = 2;
static constexpr int kN  = 32768;   // 32^3 query tokens
static constexpr int kC  = 128;
static constexpr int kNH = 4;
static constexpr int kDH = 32;      // C / NUM_HEADS
static constexpr int kM  = 512;     // 8^3 reduced KV tokens
static constexpr float kScale = 0.17677669529663687f;  // 1/sqrt(32)

// Generic pointer to a __shared__ object: low 32 bits are the LDS byte offset
// (ISA 10.2 aperture mapping).
__device__ __forceinline__ uint32_t lds_offset(const void* p) {
  return (uint32_t)(uintptr_t)p;
}
// CDNA5 async DMA: 16 bytes global -> LDS, tracked by ASYNCcnt (no VGPRs).
__device__ __forceinline__ void async_copy_b128(uint32_t lds_dst,
                                                const void* gsrc) {
  uint64_t ga = (uint64_t)(uintptr_t)gsrc;
  asm volatile("global_load_async_to_lds_b128 %0, %1, off"
               :: "v"(lds_dst), "v"(ga) : "memory");
}

// -------------------- WMMA fragment loaders (contiguous k order, same in A and B) -----------
// A (16x32 f16, MxK): lane L holds row M=L&15; K half = 16*(L>>4);
//   element j -> K = j + Khalf (contiguous 16).
__device__ __forceinline__ v16h load_a_frag(const float* A, int lda, int m0,
                                            int k0, int lane) {
  const float* p = A + (size_t)(m0 + (lane & 15)) * lda + k0 + ((lane >> 4) << 4);
  v16h a;
#pragma unroll
  for (int j = 0; j < 16; ++j) a[j] = (_Float16)p[j];
  return a;
}
__device__ __forceinline__ v16h load_a_frag(const _Float16* A, int lda, int m0,
                                            int k0, int lane) {
  const _Float16* p = A + (size_t)(m0 + (lane & 15)) * lda + k0 + ((lane >> 4) << 4);
  v16h a;
#pragma unroll
  for (int j = 0; j < 16; ++j) a[j] = p[j];
  return a;
}
// B (32x16 f16, KxN) from row-major weight W[N][K] (out = A @ W^T):
// lane L holds column n = L&15; K run = k0 + 16*(L>>4) + j (contiguous 16).
__device__ __forceinline__ v16h load_b_frag(const float* W, int ldw, int n0,
                                            int k0, int lane) {
  const float* p = W + (size_t)(n0 + (lane & 15)) * ldw + k0 + ((lane >> 4) << 4);
  v16h b;
#pragma unroll
  for (int j = 0; j < 16; ++j) b[j] = (_Float16)p[j];
  return b;
}

// -------------------- generic GEMM: out[M,N] = A[M,K] @ W[N,K]^T + bias -----
// grid = (Mrows/16, Ntot/128); block = 256 (8 waves); 1 wave = one 16x16 tile.
// The 16xK A strip is async-DMA'd to LDS once and shared by all 8 waves.
template <typename AT, bool OUT_HALF>
__global__ __launch_bounds__(256) void gemm_wmma_kernel(
    const AT* __restrict__ A, const float* __restrict__ W,
    const float* __restrict__ bias, void* __restrict__ out, int K) {
  __shared__ __align__(16) char atile[16 * 128 * 4];  // up to 8 KB
  const int lane = threadIdx.x & 31;
  const int wave = threadIdx.x >> 5;
  const int m0 = blockIdx.x * 16;
  const int n0 = (blockIdx.y * 8 + wave) * 16;
  const int Ntot = gridDim.y * 128;

  {  // cooperative async copy of the contiguous [16 x K] strip
    const int nbytes = 16 * K * (int)sizeof(AT);
    const char* gsrc = (const char*)(A + (size_t)m0 * K);
    const uint32_t lbase = lds_offset(atile);
    for (int off = threadIdx.x * 16; off < nbytes; off += 256 * 16)
      async_copy_b128(lbase + off, gsrc + off);
    asm volatile("s_wait_asynccnt 0x0" ::: "memory");
    __syncthreads();
  }
  const AT* lA = (const AT*)atile;  // LDS, row-major [16][K]

  __builtin_prefetch(W + (size_t)n0 * K, 0, 3);  // -> global_prefetch_b8

  v8f acc = {};
  for (int k0 = 0; k0 < K; k0 += 32) {
    v16h a = load_a_frag(lA, K, 0, k0, lane);
    v16h b = load_b_frag(W, K, n0, k0, lane);
    acc = WMMA_F16(a, b, acc);
  }
  // C/D layout: VGPR r, lanes<16 -> row r, lanes>=16 -> row r+8; col = lane&15
  const int col  = n0 + (lane & 15);
  const int row0 = m0 + ((lane >> 4) << 3);
  const float bv = bias[col];
  if (OUT_HALF) {
    _Float16* o = (_Float16*)out;
#pragma unroll
    for (int r = 0; r < 8; ++r)
      o[(size_t)(row0 + r) * Ntot + col] = (_Float16)(acc[r] + bv);
  } else {
    float* o = (float*)out;
#pragma unroll
    for (int r = 0; r < 8; ++r)
      o[(size_t)(row0 + r) * Ntot + col] = acc[r] + bv;
  }
}

// -------------------- SR conv (4x4x4 stride-4) as implicit GEMM ------------
// out[B*M=1024, C=128], K = C*64 = 8192 ; k = s*128 + ci, s = i*16+j*4+l.
// The gathered 16x32 patch chunk (identical for all 8 waves) is double-
// buffered in LDS via async DMA: chunk i+1 streams in while chunk i is in the
// WMMA pipe. Per-iteration ordering: wait asynccnt 0 (chunk i landed) ->
// wait dscnt 0 (my reads of chunk i-1 done) -> barrier -> issue i+1 -> compute.
__global__ __launch_bounds__(256) void conv_sr_kernel(
    const float* __restrict__ x, const float* __restrict__ w,
    const float* __restrict__ bias, float* __restrict__ out) {
  __shared__ __align__(16) float achunk[2][16 * 32];  // 2 x 2 KB
  const int tid = threadIdx.x;
  const int lane = tid & 31;
  const int wave = tid >> 5;

  // copy-thread mapping: 128 threads, one b128 per (row, 16B segment)
  const int crow = tid >> 3;       // 0..31 (only <16 used)
  const int cseg = (tid & 7) * 4;  // float offset within 32-float chunk
  size_t cbase = 0;
  if (tid < 128) {
    int rr = blockIdx.x * 16 + crow;
    int bb = rr >> 9, mm = rr & 511;
    int dd = mm >> 6, hh = (mm >> 3) & 7, ww = mm & 7;
    cbase = ((size_t)bb * kN + (size_t)(4 * dd) * 1024 + (4 * hh) * 32 +
             (4 * ww)) * kC;
  }
  auto issue_chunk = [&](int k0, int buf) {
    if (tid < 128) {
      int s = k0 >> 7, ci = k0 & 127;
      int i = s >> 4, jj = (s >> 2) & 3, l = s & 3;
      const float* src =
          x + cbase + ((size_t)i * 1024 + jj * 32 + l) * kC + ci + cseg;
      async_copy_b128(lds_offset(&achunk[buf][crow * 32 + cseg]), src);
    }
  };

  const int n0 = wave * 16;
  const int kh = (lane >> 4) << 4;
  v8f acc = {};
  issue_chunk(0, 0);
  for (int k0 = 0; k0 < 8192; k0 += 32) {
    const int cur = (k0 >> 5) & 1;
    asm volatile("s_wait_asynccnt 0x0" ::: "memory");  // chunk cur landed
    asm volatile("s_wait_dscnt 0x0" ::: "memory");     // my old reads done
    __syncthreads();
    if (k0 + 32 < 8192) issue_chunk(k0 + 32, cur ^ 1);

    // A fragment from the staged LDS chunk [16][32]
    const float* lc = &achunk[cur][0];
    v16h a;
#pragma unroll
    for (int j = 0; j < 16; ++j)
      a[j] = (_Float16)lc[(lane & 15) * 32 + kh + j];

    // B fragment from sr_w flat [co][ci][s] = co*8192 + ci*64 + s
    int kb = k0 + ((lane >> 4) << 4);
    const float* p = w + (size_t)(n0 + (lane & 15)) * 8192 +
                     (size_t)(kb & 127) * 64 + (kb >> 7);
    v16h bf;
#pragma unroll
    for (int j = 0; j < 16; ++j) bf[j] = (_Float16)p[j * 64];

    acc = WMMA_F16(a, bf, acc);
  }
  const int col  = n0 + (lane & 15);
  const int row0 = blockIdx.x * 16 + ((lane >> 4) << 3);
  const float bv = bias[col];
#pragma unroll
  for (int r = 0; r < 8; ++r) out[(size_t)(row0 + r) * kC + col] = acc[r] + bv;
}

// -------------------- LayerNorm over C=128, one wave per row ---------------
__global__ __launch_bounds__(256) void ln_kernel(
    const float* __restrict__ xin, const float* __restrict__ g,
    const float* __restrict__ bt, _Float16* __restrict__ xout) {
  const int lane = threadIdx.x & 31;
  const int row = blockIdx.x * 8 + (threadIdx.x >> 5);
  const float* p = xin + (size_t)row * kC;
  float v[4];
#pragma unroll
  for (int i = 0; i < 4; ++i) v[i] = p[lane * 4 + i];
  float s = v[0] + v[1] + v[2] + v[3];
#pragma unroll
  for (int msk = 16; msk >= 1; msk >>= 1) s += __shfl_xor(s, msk, 32);
  const float mu = s * (1.0f / 128.0f);
  float q = 0.f;
#pragma unroll
  for (int i = 0; i < 4; ++i) { float d = v[i] - mu; q += d * d; }
#pragma unroll
  for (int msk = 16; msk >= 1; msk >>= 1) q += __shfl_xor(q, msk, 32);
  const float inv = rsqrtf(q * (1.0f / 128.0f) + 1e-5f);
#pragma unroll
  for (int i = 0; i < 4; ++i) {
    int c = lane * 4 + i;
    xout[(size_t)row * kC + c] = (_Float16)((v[i] - mu) * inv * g[c] + bt[c]);
  }
}

// -------------------- repack KV: [B*M, 256] -> K[B,h,M,dh], V^T[B,h,dh,M] ---
__global__ __launch_bounds__(256) void repack_kv_kernel(
    const _Float16* __restrict__ kv, _Float16* __restrict__ kb,
    _Float16* __restrict__ vt) {
  const int idx = blockIdx.x * blockDim.x + threadIdx.x;  // 1024*256
  const int rowm = idx >> 8, n = idx & 255;
  const int b = rowm >> 9, m = rowm & 511;
  const _Float16 val = kv[idx];
  if (n < 128) {
    int h = n >> 5, d = n & 31;
    kb[((size_t)(b * 4 + h) * kM + m) * kDH + d] = val;
  } else {
    int nn = n - 128, h = nn >> 5, d = nn & 31;
    vt[((size_t)(b * 4 + h) * kDH + d) * kM + m] = val;
  }
}

// -------------------- flash attention, 1 wave = 16 query rows of one head ---
__global__ __launch_bounds__(256) void attn_kernel(
    const _Float16* __restrict__ q, const _Float16* __restrict__ kb,
    const _Float16* __restrict__ vt, _Float16* __restrict__ obuf) {
  __shared__ __align__(16) _Float16 psc[8][16 * 32];  // 1 KB / wave
  const int lane = threadIdx.x & 31;
  const int wave = threadIdx.x >> 5;
  const int g = blockIdx.x * 8 + wave;  // 0..16383
  const int ntile = g & 2047, h = (g >> 11) & 3, b = g >> 13;
  const int n0 = ntile * 16;

  // Q A-fragment: K-dim = dh = 32 -> exactly one wmma per 16x16 S tile.
  const v16h qa = load_a_frag(q, kC, b * kN + n0, h * kDH, lane);
  const _Float16* kbase = kb + (size_t)(b * 4 + h) * kM * kDH;
  const _Float16* vbase = vt + (size_t)(b * 4 + h) * kDH * kM;

  v8f o0 = {}, o1 = {};
  float mrun[8], lrun[8];
#pragma unroll
  for (int r = 0; r < 8; ++r) { mrun[r] = -1e30f; lrun[r] = 0.f; }

  for (int mc = 0; mc < kM; mc += 32) {
    // K^T B-fragments: B[d][m], k[m][d] contiguous in d -> 32B runs per lane.
    v16h kb0, kb1;
    {
      const _Float16* p0 =
          kbase + (size_t)(mc + (lane & 15)) * kDH + ((lane >> 4) << 4);
      const _Float16* p1 = p0 + 16 * kDH;
#pragma unroll
      for (int j = 0; j < 16; ++j) { kb0[j] = p0[j]; kb1[j] = p1[j]; }
    }
    v8f s0 = {}, s1 = {};
    s0 = WMMA_F16(qa, kb0, s0);
    s1 = WMMA_F16(qa, kb1, s1);

    // Streaming softmax update (rows = VGPR index; cols = 16 lanes per half).
    float alpha[8];
#pragma unroll
    for (int r = 0; r < 8; ++r) {
      float a0 = s0[r] * kScale, a1 = s1[r] * kScale;
      float rm = fmaxf(a0, a1);
#pragma unroll
      for (int msk = 8; msk >= 1; msk >>= 1) rm = fmaxf(rm, __shfl_xor(rm, msk, 32));
      const float mnew = fmaxf(mrun[r], rm);
      alpha[r] = __expf(mrun[r] - mnew);
      const float p0 = __expf(a0 - mnew), p1 = __expf(a1 - mnew);
      float rs = p0 + p1;
#pragma unroll
      for (int msk = 8; msk >= 1; msk >>= 1) rs += __shfl_xor(rs, msk, 32);
      lrun[r] = lrun[r] * alpha[r] + rs;
      mrun[r] = mnew;
      s0[r] = p0; s1[r] = p1;
    }
#pragma unroll
    for (int r = 0; r < 8; ++r) { o0[r] *= alpha[r]; o1[r] *= alpha[r]; }

    // Transpose P: C/D layout -> A layout through LDS (col-major [32][16]).
    {
      _Float16* base = &psc[wave][0];
      const int rowb = (lane >> 4) << 3;
      const int c0 = lane & 15;
      v8h t0, t1;
#pragma unroll
      for (int r = 0; r < 8; ++r) {
        t0[r] = (_Float16)s0[r];
        t1[r] = (_Float16)s1[r];
      }
      *(v8h*)&base[c0 * 16 + rowb] = t0;          // 16B aligned ds_store
      *(v8h*)&base[(c0 + 16) * 16 + rowb] = t1;
    }
    asm volatile("s_wait_dscnt 0x0" ::: "memory");  // same-wave LDS RAW fence
    v16h pa;
    {
      const int m = lane & 15, khp = (lane >> 4) << 4;
#pragma unroll
      for (int j = 0; j < 16; ++j)
        pa[j] = psc[wave][(khp + j) * 16 + m];
    }

    // V B-fragments from V^T[B,h,dh,M]: contiguous in m -> 32B runs per lane.
    v16h vb0, vb1;
    {
      const _Float16* pv =
          vbase + (size_t)(lane & 15) * kM + mc + ((lane >> 4) << 4);
#pragma unroll
      for (int j = 0; j < 16; ++j) { vb0[j] = pv[j]; vb1[j] = pv[16 * kM + j]; }
    }
    o0 = WMMA_F16(pa, vb0, o0);
    o1 = WMMA_F16(pa, vb1, o1);
  }

  // Epilogue: divide by row sum, store f16 into obuf[B*N, 128].
  const int rowb = (lane >> 4) << 3;
  const int c = lane & 15;
#pragma unroll
  for (int r = 0; r < 8; ++r) {
    const float inv = 1.0f / lrun[r];
    const size_t rw = (size_t)(b * kN + n0 + rowb + r) * kC + h * kDH;
    obuf[rw + c] = (_Float16)(o0[r] * inv);
    obuf[rw + 16 + c] = (_Float16)(o1[r] * inv);
  }
}

// ---------------------------------------------------------------------------
extern "C" void kernel_launch(void* const* d_in, const int* in_sizes, int n_in,
                              void* d_out, int out_size, void* d_ws,
                              size_t ws_size, hipStream_t stream) {
  const float* x   = (const float*)d_in[0];
  const float* Wq  = (const float*)d_in[1];
  const float* bq  = (const float*)d_in[2];
  const float* Wkv = (const float*)d_in[3];
  const float* bkv = (const float*)d_in[4];
  const float* srw = (const float*)d_in[5];
  const float* srb = (const float*)d_in[6];
  const float* lng = (const float*)d_in[7];
  const float* lnb = (const float*)d_in[8];
  const float* Wp  = (const float*)d_in[9];
  const float* bp  = (const float*)d_in[10];

  char* ws = (char*)d_ws;
  float*    conv_out = (float*)(ws + 0);          //  512 KB: [1024,128] f32
  _Float16* xln      = (_Float16*)(ws + 524288);  //  256 KB: [1024,128] f16
  _Float16* kvb      = (_Float16*)(ws + 786432);  //  512 KB: [1024,256] f16
  _Float16* kbuf     = (_Float16*)(ws + 1310720); //  256 KB: [B,h,M,dh] f16
  _Float16* vbt      = (_Float16*)(ws + 1572864); //  256 KB: [B,h,dh,M] f16
  _Float16* qb       = (_Float16*)(ws + 1835008); //   16 MB: [B*N,128] f16
  _Float16* ob       = (_Float16*)(ws + 18612224);//   16 MB: [B*N,128] f16
  float* out = (float*)d_out;

  // 1) spatial-reduction conv (implicit GEMM, K=8192, async double-buffer)
  conv_sr_kernel<<<64, 256, 0, stream>>>(x, srw, srb, conv_out);
  // 2) layernorm -> f16
  ln_kernel<<<128, 256, 0, stream>>>(conv_out, lng, lnb, xln);
  // 3) KV projection [1024,256]
  gemm_wmma_kernel<_Float16, true>
      <<<dim3(64, 2), 256, 0, stream>>>(xln, Wkv, bkv, kvb, 128);
  // 4) repack K / V^T for attention fragment loads
  repack_kv_kernel<<<1024, 256, 0, stream>>>(kvb, kbuf, vbt);
  // 5) Q projection [65536,128] -> f16
  gemm_wmma_kernel<float, true>
      <<<dim3(4096, 1), 256, 0, stream>>>(x, Wq, bq, qb, 128);
  // 6) flash attention
  attn_kernel<<<2048, 256, 0, stream>>>(qb, kbuf, vbt, ob);
  // 7) output projection -> f32 d_out
  gemm_wmma_kernel<_Float16, false>
      <<<dim3(4096, 1), 256, 0, stream>>>(ob, Wp, bp, out, 128);
}
